// DiffPoolLayer_32384053412284
// MI455X (gfx1250) — compile-verified, matrix operand order checked by reading, not run on hardware
//
#include <hip/hip_runtime.h>
#include <hip/hip_bf16.h>

typedef __attribute__((ext_vector_type(16))) __bf16 v16bf;
typedef __attribute__((ext_vector_type(8)))  float  v8f;
typedef __attribute__((ext_vector_type(4)))  int    v4i_t;

#define N_NODES 16384
#define N_EDGES 524288
#define IN_CH   128
#define NCAT    256   // HID (128) + NCLUST (128) fused
#define HID     128
#define NCLUST  128

// ---------------- gfx1250 async global->LDS staging (ASYNCcnt path) ----------
#if defined(__has_builtin)
#  if __has_builtin(__builtin_amdgcn_global_load_async_to_lds_b128) && \
      __has_builtin(__builtin_amdgcn_s_wait_asynccnt)
#    define USE_ASYNC_LDS 1
#  endif
#endif
#ifndef USE_ASYNC_LDS
#  define USE_ASYNC_LDS 0
#endif

#define AS1 __attribute__((address_space(1)))
#define AS3 __attribute__((address_space(3)))

// Copy 16 bytes (8 bf16) global -> LDS. Flat LDS aperture maps addr[31:0] to
// the LDS byte offset, so the int round-trip yields a valid AS(3) pointer.
// The builtin's params are pointers to int32x4 in AS1/AS3.
__device__ __forceinline__ void stage16(__bf16* lds_dst, const __bf16* g_src) {
#if USE_ASYNC_LDS
  AS3 v4i_t* l =
      (AS3 v4i_t*)(unsigned long long)(unsigned)(unsigned long long)lds_dst;
  AS1 v4i_t* g = (AS1 v4i_t*)(unsigned long long)g_src;
  __builtin_amdgcn_global_load_async_to_lds_b128(g, l, 0, 0);
#else
  *(float4*)lds_dst = *(const float4*)g_src;
#endif
}
__device__ __forceinline__ void stage_wait() {
#if USE_ASYNC_LDS
  __builtin_amdgcn_s_wait_asynccnt(0);
#endif
  __syncthreads();
}

// ---------------------------------------------------------------- fills
__global__ void k_fill_f32(float* __restrict__ p, float v, int n) {
  int i = blockIdx.x * blockDim.x + threadIdx.x;
  int st = gridDim.x * blockDim.x;
  for (; i < n; i += st) p[i] = v;
}
__global__ void k_fill_u32(unsigned int* __restrict__ p, unsigned int v, int n) {
  int i = blockIdx.x * blockDim.x + threadIdx.x;
  int st = gridDim.x * blockDim.x;
  for (; i < n; i += st) p[i] = v;
}

// ------------------------------------------- fused GEMM: Hcat = x @ [Wemb|Wasn]
// One wave per 16x16 output tile; K=128 swept in 4 chunks of 32 via
// v_wmma_f32_16x16x32_bf16 (f32 accumulate).
__global__ void k_gemm_xw(const float* __restrict__ x,
                          const float* __restrict__ Wemb,
                          const float* __restrict__ Wasn,
                          float* __restrict__ Hcat) {
  const int wave = threadIdx.x >> 5;
  const int lane = threadIdx.x & 31;
  const int tile = blockIdx.x * 8 + wave;       // 1024 m-tiles * 16 n-tiles
  const int mt = tile >> 4;
  const int nt = tile & 15;
  const int m0 = mt << 4;
  const int h  = lane >> 4;
  const int l16 = lane & 15;
  const int row = m0 + l16;
  const float* Wp = (nt < 8) ? Wemb : Wasn;
  const int ncol = ((nt & 7) << 4) + l16;       // column within 128-wide W

  v8f acc = {};
#pragma unroll
  for (int kc = 0; kc < 4; ++kc) {
    const int kb = kc << 5;
    v16bf a, b;
#pragma unroll
    for (int g = 0; g < 2; ++g) {
      const int k = kb + g * 16 + h * 8;
      const float4 p0 = *(const float4*)(x + row * IN_CH + k);
      const float4 p1 = *(const float4*)(x + row * IN_CH + k + 4);
      a[g*8+0] = (__bf16)p0.x; a[g*8+1] = (__bf16)p0.y;
      a[g*8+2] = (__bf16)p0.z; a[g*8+3] = (__bf16)p0.w;
      a[g*8+4] = (__bf16)p1.x; a[g*8+5] = (__bf16)p1.y;
      a[g*8+6] = (__bf16)p1.z; a[g*8+7] = (__bf16)p1.w;
#pragma unroll
      for (int j = 0; j < 8; ++j)
        b[g*8+j] = (__bf16)Wp[(k + j) * 128 + ncol];
    }
    acc = __builtin_amdgcn_wmma_f32_16x16x32_bf16(false, a, false, b,
                                                  (short)0, acc, false, false);
  }
#pragma unroll
  for (int r = 0; r < 8; ++r)
    Hcat[(m0 + r + 8 * h) * NCAT + (nt << 4) + l16] = acc[r];
}

// ---------------------------------------------------------------- degrees
__global__ void k_edge_deg(const int* __restrict__ dst, float* __restrict__ deg) {
  const int e = blockIdx.x * blockDim.x + threadIdx.x;
  if (e < N_EDGES) unsafeAtomicAdd(deg + dst[e], 1.0f);
}
__global__ void k_dinv(const float* __restrict__ deg, float* __restrict__ dinv) {
  const int i = blockIdx.x * blockDim.x + threadIdx.x;
  if (i < N_NODES) dinv[i] = rsqrtf(deg[i]);
}

// ----------------------------------------- dedup: A[src,dst] is SET to 1
__global__ void k_dedup(const int* __restrict__ src, const int* __restrict__ dst,
                        unsigned int* __restrict__ bitmap,
                        unsigned char* __restrict__ flags) {
  const int e = blockIdx.x * blockDim.x + threadIdx.x;
  if (e >= N_EDGES) return;
  const unsigned int idx = (unsigned int)src[e] * (unsigned int)N_NODES +
                           (unsigned int)dst[e];
  const unsigned int bit = 1u << (idx & 31u);
  const unsigned int old = atomicOr(bitmap + (idx >> 5), bit);
  flags[e] = (old & bit) ? 0u : 1u;   // first arrival owns the edge
}

// ------------------------- normalized edge scatter: agg[dst] += w * h[src]
__global__ void k_scatter(const float* __restrict__ Hcat,
                          const float* __restrict__ dinv,
                          const int* __restrict__ src, const int* __restrict__ dst,
                          float* __restrict__ agg) {
  const int t = blockIdx.x * blockDim.x + threadIdx.x;
  const int e = t >> 5;
  const int c = (t & 31) << 3;
  const int s = src[e], d = dst[e];
  const float w = dinv[s] * dinv[d];
  const float4 h0 = *(const float4*)(Hcat + s * NCAT + c);
  const float4 h1 = *(const float4*)(Hcat + s * NCAT + c + 4);
  float* o = agg + d * NCAT + c;
  unsafeAtomicAdd(o + 0, h0.x * w); unsafeAtomicAdd(o + 1, h0.y * w);
  unsafeAtomicAdd(o + 2, h0.z * w); unsafeAtomicAdd(o + 3, h0.w * w);
  unsafeAtomicAdd(o + 4, h1.x * w); unsafeAtomicAdd(o + 5, h1.y * w);
  unsafeAtomicAdd(o + 6, h1.z * w); unsafeAtomicAdd(o + 7, h1.w * w);
}

// -------- finalize: add self-loop term + bias, ReLU half -> bf16 x_emb,
// softmax half -> f32 S (output) + bf16 S (for pooling GEMMs)
__global__ void k_finalize(const float* __restrict__ Hcat,
                           const float* __restrict__ agg,
                           const float* __restrict__ dinv,
                           const float* __restrict__ bemb,
                           const float* __restrict__ basn,
                           float* __restrict__ outS,
                           __bf16* __restrict__ Sb,
                           __bf16* __restrict__ Xb) {
  const int i = blockIdx.x;
  const int c = threadIdx.x;                 // 0..255
  const float di = dinv[i];
  const float selfw = di * di;
  const float val = agg[i * NCAT + c] + Hcat[i * NCAT + c] * selfw;

  __shared__ float lg[128], ex[128], rd[128];
  if (c < 128) {
    const float ve = fmaxf(val + bemb[c], 0.0f);
    Xb[i * HID + c] = (__bf16)ve;
  } else {
    lg[c - 128] = val + basn[c - 128];
  }
  __syncthreads();
  if (c < 128) rd[c] = lg[c];
  __syncthreads();
  for (int s2 = 64; s2 > 0; s2 >>= 1) {
    if (c < s2) rd[c] = fmaxf(rd[c], rd[c + s2]);
    __syncthreads();
  }
  const float mx = rd[0];
  __syncthreads();
  if (c < 128) { const float e2 = __expf(lg[c] - mx); ex[c] = e2; rd[c] = e2; }
  __syncthreads();
  for (int s2 = 64; s2 > 0; s2 >>= 1) {
    if (c < s2) rd[c] += rd[c + s2];
    __syncthreads();
  }
  const float inv = 1.0f / rd[0];
  if (c < 128) {
    const float sv = ex[c] * inv;
    outS[(size_t)i * NCLUST + c] = sv;
    Sb[i * NCLUST + c] = (__bf16)sv;
  }
}

// ------------- x_pooled = S^T @ x_emb : split-K WMMA, 32 k-rows per chunk.
// Chunk rows of S and x_emb are staged once per block into LDS (async path on
// gfx1250), then all 8 waves build fragments from LDS instead of re-reading L2.
__global__ void k_pool_x(const __bf16* __restrict__ Sb,
                         const __bf16* __restrict__ Xb,
                         float* __restrict__ out) {
  const int t    = threadIdx.x;
  const int wave = t >> 5;
  const int lane = t & 31;
  const int h = lane >> 4, l16 = lane & 15;
  const int m = (wave << 4) + l16;            // output row this lane feeds A with
  const int seg = t & 15;                     // 16B segment within a 256B row
  const int rp  = t >> 4;                     // 0..15 (stages rows rp and rp+16)

  __shared__ __bf16 Ush[32 * 128];            // S rows of this chunk
  __shared__ __bf16 Vsh[32 * 128];            // x_emb rows of this chunk

  v8f acc[8] = {};
  const int c0 = blockIdx.x * 8;              // 64 blocks * 8 chunks = 512 chunks
  for (int ci = 0; ci < 8; ++ci) {
    const int kb = (c0 + ci) << 5;
    __syncthreads();                          // previous chunk's reads done
    stage16(Ush + rp * 128 + seg * 8,        Sb + (size_t)(kb + rp) * 128 + seg * 8);
    stage16(Ush + (rp + 16) * 128 + seg * 8, Sb + (size_t)(kb + rp + 16) * 128 + seg * 8);
    stage16(Vsh + rp * 128 + seg * 8,        Xb + (size_t)(kb + rp) * 128 + seg * 8);
    stage16(Vsh + (rp + 16) * 128 + seg * 8, Xb + (size_t)(kb + rp + 16) * 128 + seg * 8);
    stage_wait();

    v16bf a;
#pragma unroll
    for (int e = 0; e < 16; ++e) {
      const int kl = ((e >> 3) << 4) + (h << 3) + (e & 7);
      a[e] = Ush[kl * 128 + m];               // A = S^T  (A[m][k] = S[k][m])
    }
#pragma unroll
    for (int nt = 0; nt < 8; ++nt) {
      v16bf b;
#pragma unroll
      for (int e = 0; e < 16; ++e) {
        const int kl = ((e >> 3) << 4) + (h << 3) + (e & 7);
        b[e] = Vsh[kl * 128 + (nt << 4) + l16];
      }
      acc[nt] = __builtin_amdgcn_wmma_f32_16x16x32_bf16(false, a, false, b,
                                                        (short)0, acc[nt],
                                                        false, false);
    }
  }
#pragma unroll
  for (int nt = 0; nt < 8; ++nt)
#pragma unroll
    for (int r = 0; r < 8; ++r)
      unsafeAtomicAdd(out + ((wave << 4) + r + 8 * h) * HID + (nt << 4) + l16,
                      acc[nt][r]);
}

// ------- A_pooled = sum over (deduped) edges of S[src]^T (x) S[dst],
// batched 32 edges per WMMA chunk. The 32 gathered src/dst rows of S are
// staged into LDS once per block per chunk.
__global__ void k_pool_A(const __bf16* __restrict__ Sb,
                         const int* __restrict__ src, const int* __restrict__ dst,
                         const unsigned char* __restrict__ flags,
                         float* __restrict__ out) {
  const int t    = threadIdx.x;
  const int wave = t >> 5;
  const int lane = t & 31;
  const int h = lane >> 4, l16 = lane & 15;
  const int m = (wave << 4) + l16;
  const int seg = t & 15;
  const int rp  = t >> 4;                     // 0..15

  __shared__ __bf16 Ush[32 * 128];            // S[src] rows
  __shared__ __bf16 Vsh[32 * 128];            // S[dst] rows
  __shared__ unsigned char Fsh[32];           // dedup flags

  v8f acc[8] = {};
  const int c0 = blockIdx.x * 64;             // 256 blocks * 64 chunks = 16384
  for (int ci = 0; ci < 64; ++ci) {
    const int kb = (c0 + ci) << 5;
    __syncthreads();
    const int e0 = kb + rp, e1 = kb + rp + 16;
    const int su0 = src[e0], su1 = src[e1];
    const int dv0 = dst[e0], dv1 = dst[e1];
    stage16(Ush + rp * 128 + seg * 8,        Sb + (size_t)su0 * NCLUST + seg * 8);
    stage16(Ush + (rp + 16) * 128 + seg * 8, Sb + (size_t)su1 * NCLUST + seg * 8);
    stage16(Vsh + rp * 128 + seg * 8,        Sb + (size_t)dv0 * NCLUST + seg * 8);
    stage16(Vsh + (rp + 16) * 128 + seg * 8, Sb + (size_t)dv1 * NCLUST + seg * 8);
    if (t < 32) Fsh[t] = flags[kb + t];
    stage_wait();

    v16bf a;
#pragma unroll
    for (int e = 0; e < 16; ++e) {
      const int kl = ((e >> 3) << 4) + (h << 3) + (e & 7);
      a[e] = Fsh[kl] ? Ush[kl * 128 + m] : (__bf16)0.0f;   // dedup weight
    }
#pragma unroll
    for (int nt = 0; nt < 8; ++nt) {
      v16bf b;
#pragma unroll
      for (int e = 0; e < 16; ++e) {
        const int kl = ((e >> 3) << 4) + (h << 3) + (e & 7);
        b[e] = Vsh[kl * 128 + (nt << 4) + l16];
      }
      acc[nt] = __builtin_amdgcn_wmma_f32_16x16x32_bf16(false, a, false, b,
                                                        (short)0, acc[nt],
                                                        false, false);
    }
  }
#pragma unroll
  for (int nt = 0; nt < 8; ++nt)
#pragma unroll
    for (int r = 0; r < 8; ++r)
      unsafeAtomicAdd(out + ((wave << 4) + r + 8 * h) * NCLUST + (nt << 4) + l16,
                      acc[nt][r]);
}

// ---------------------------------------------------------------- launch
extern "C" void kernel_launch(void* const* d_in, const int* in_sizes, int n_in,
                              void* d_out, int out_size, void* d_ws, size_t ws_size,
                              hipStream_t stream) {
  const float* x    = (const float*)d_in[0];
  const int*   ei   = (const int*)d_in[1];
  const float* Wemb = (const float*)d_in[2];
  const float* bemb = (const float*)d_in[3];
  const float* Wasn = (const float*)d_in[4];
  const float* basn = (const float*)d_in[5];
  const int* src = ei;
  const int* dst = ei + N_EDGES;

  char* ws = (char*)d_ws;
  size_t off = 0;
  float* Hcat = (float*)(ws + off); off += (size_t)N_NODES * NCAT * 4;
  float* agg  = (float*)(ws + off); off += (size_t)N_NODES * NCAT * 4;
  float* deg  = (float*)(ws + off); off += (size_t)N_NODES * 4;
  float* dinv = (float*)(ws + off); off += (size_t)N_NODES * 4;
  __bf16* Sb  = (__bf16*)(ws + off); off += (size_t)N_NODES * NCLUST * 2;
  __bf16* Xb  = (__bf16*)(ws + off); off += (size_t)N_NODES * HID * 2;
  unsigned char* flags = (unsigned char*)(ws + off); off += (size_t)N_EDGES;
  const size_t bm_words = (size_t)N_NODES * N_NODES / 32;
  unsigned int* bitmap = (unsigned int*)(ws + off);
  const bool dedup = (off + bm_words * 4) <= ws_size;

  float* outx = (float*)d_out;          // x_pooled  [128x128]
  float* outA = outx + 16384;           // A_pooled  [128x128]
  float* outS = outx + 32768;           // S         [16384x128]

  k_fill_f32<<<1024, 256, 0, stream>>>(agg, 0.0f, N_NODES * NCAT);
  k_fill_f32<<<32,   256, 0, stream>>>(outx, 0.0f, 32768);
  k_fill_f32<<<64,   256, 0, stream>>>(deg, 1.0f, N_NODES);   // self-loop degree
  k_fill_u32<<<512,  256, 0, stream>>>((unsigned int*)flags, 0x01010101u,
                                       N_EDGES / 4);
  k_gemm_xw<<<2048, 256, 0, stream>>>(x, Wemb, Wasn, Hcat);
  k_edge_deg<<<N_EDGES / 256, 256, 0, stream>>>(dst, deg);
  k_dinv<<<N_NODES / 256, 256, 0, stream>>>(deg, dinv);
  if (dedup) {
    k_fill_u32<<<4096, 256, 0, stream>>>(bitmap, 0u, (int)bm_words);
    k_dedup<<<N_EDGES / 256, 256, 0, stream>>>(src, dst, bitmap, flags);
  }
  k_scatter<<<(N_EDGES * 32) / 256, 256, 0, stream>>>(Hcat, dinv, src, dst, agg);
  k_finalize<<<N_NODES, 256, 0, stream>>>(Hcat, agg, dinv, bemb, basn,
                                          outS, Sb, Xb);
  k_pool_x<<<64,  256, 0, stream>>>(Sb, Xb, outx);
  k_pool_A<<<256, 256, 0, stream>>>(Sb, src, dst, flags, outA);
}
